// BigramLanguageModel_75067438400017
// MI455X (gfx1250) — matrix operand order repaired and checked by reading, not compiled, not run on hardware
//
#include <hip/hip_runtime.h>

#define BB   1024
#define TT   128
#define DD   128
#define HH   4
#define HSZ  32
#define VV   256
#define EE   8
#define KTOP 2
#define FF   512
#define LL   6
#define BT   (BB*TT)

typedef __bf16 v16bf __attribute__((ext_vector_type(16)));
typedef float  v8f   __attribute__((ext_vector_type(8)));
typedef int    v4i   __attribute__((vector_size(16)));   // matches async builtin pointee

union BFrag { v16bf v; uint4 q[2]; };

// Async-to-LDS path (CDNA5): guarded so compile stays green on either toolchain.
#if defined(__HIP_DEVICE_COMPILE__) && __has_builtin(__builtin_amdgcn_global_load_async_to_lds_b128)
#define USE_ASYNC 1
#else
#define USE_ASYNC 0
#endif

template<int N>
__device__ __forceinline__ void wait_async() {
#if __has_builtin(__builtin_amdgcn_s_wait_asynccnt)
  __builtin_amdgcn_s_wait_asynccnt((short)N);
#else
  if constexpr (N == 0) asm volatile("s_wait_asynccnt 0x0" ::: "memory");
  else                  asm volatile("s_wait_asynccnt 0x6" ::: "memory");
#endif
}

__device__ __forceinline__ unsigned short f2bf(float f) {
  union { float f; unsigned u; } x; x.f = f;
  unsigned r = x.u + 0x7FFFu + ((x.u >> 16) & 1u);
  return (unsigned short)(r >> 16);
}
__device__ __forceinline__ float bf2f(unsigned short s) {
  union { unsigned u; float f; } x; x.u = ((unsigned)s) << 16; return x.f;
}
__device__ __forceinline__ float gelu_exact(float x) {
  return 0.5f * x * (1.0f + erff(x * 0.70710678118654752f));
}
__device__ __forceinline__ v16bf ld_frag(const unsigned short* p0, const unsigned short* p1) {
  BFrag u;
  u.q[0] = *(const uint4*)p0;
  u.q[1] = *(const uint4*)p1;
  return u.v;
}
__device__ __forceinline__ v8f wmma_bf16(v16bf a, v16bf b, v8f c) {
  return __builtin_amdgcn_wmma_f32_16x16x32_bf16(false, a, false, b, (short)0, c, false, false);
}

// ---------------------------------------------------------------------------
// bf16 WMMA GEMM: C[MxN] = A[MxK] * Bt^T, with B PRE-TRANSPOSED as Bt[N][K].
// Block = 128 threads = 4 waves (2 M x 2 N); block tile 64x128, K-step 32.
// Each wave: 32x64 = 2x4 16x16 tiles -> 8 WMMAs per K-step.
// Double-buffered LDS staging; async GLOBAL_LOAD_ASYNC_TO_LDS_B128 when
// available (ASYNCcnt-synchronized), plain copies otherwise.
// A frag: lane m=l16, k = 16*half+i -> two b128 loads.
// B frag: lane n=l16, k = 16*half+i -> two b128 loads from sB[n][k].
// EPI: 1 store bf16 | 2 resid+bias f32 | 3 bias+gelu bf16 | 4 x += s[m]*(v+bias) | 5 bias f32
// ---------------------------------------------------------------------------
template<int EPI>
__global__ __launch_bounds__(128)
void gemm_bf16_kernel(const unsigned short* __restrict__ A,
                      const unsigned short* __restrict__ Bt,
                      const float* __restrict__ bias,
                      const float* __restrict__ scale, int scale_stride,
                      float* __restrict__ Cf,
                      unsigned short* __restrict__ Cb,
                      const float* __restrict__ resid,
                      int M, int N, int Kd)
{
  __shared__ __align__(16) unsigned short sA[2][64][32];   // 8 KB
  __shared__ __align__(16) unsigned short sB[2][128][32];  // 16 KB
  const int tid  = threadIdx.x;
  const int wave = tid >> 5, lane = tid & 31;
  const int wm = wave >> 1, wn = wave & 1;
  const int half = lane >> 4, l16 = lane & 15;
  const int bm = blockIdx.x, bn = blockIdx.y;
  const int kq = Kd >> 3;  // uint4 per K-row

  const v8f vz = {0.f,0.f,0.f,0.f,0.f,0.f,0.f,0.f};
  v8f acc[2][4] = {{vz, vz, vz, vz}, {vz, vz, vz, vz}};

  const uint4* Agb = (const uint4*)(A  + (size_t)bm * 64  * Kd);
  const uint4* Bgb = (const uint4*)(Bt + (size_t)bn * 128 * Kd);

  auto stage = [&](int kt, int buf) {
    const int kc = kt * 4;  // uint4 column offset of this K-step
#if USE_ASYNC
#pragma unroll
    for (int i = 0; i < 2; i++) {
      int idx = tid + i * 128;
      int r = idx >> 2, c = idx & 3;
      __builtin_amdgcn_global_load_async_to_lds_b128(
          (__attribute__((address_space(1))) v4i*)(Agb + (size_t)r * kq + kc + c),
          (__attribute__((address_space(3))) v4i*)&sA[buf][r][c * 8], 0, 0);
    }
#pragma unroll
    for (int i = 0; i < 4; i++) {
      int idx = tid + i * 128;
      int r = idx >> 2, c = idx & 3;
      __builtin_amdgcn_global_load_async_to_lds_b128(
          (__attribute__((address_space(1))) v4i*)(Bgb + (size_t)r * kq + kc + c),
          (__attribute__((address_space(3))) v4i*)&sB[buf][r][c * 8], 0, 0);
    }
#else
#pragma unroll
    for (int i = 0; i < 2; i++) {
      int idx = tid + i * 128;
      int r = idx >> 2, c = idx & 3;
      *(uint4*)&sA[buf][r][c * 8] = Agb[(size_t)r * kq + kc + c];
    }
#pragma unroll
    for (int i = 0; i < 4; i++) {
      int idx = tid + i * 128;
      int r = idx >> 2, c = idx & 3;
      *(uint4*)&sB[buf][r][c * 8] = Bgb[(size_t)r * kq + kc + c];
    }
#endif
  };

  const int nk = Kd >> 5;
  stage(0, 0);
  for (int kt = 0; kt < nk; kt++) {
    if (kt + 1 < nk) stage(kt + 1, (kt + 1) & 1);
#if USE_ASYNC
    if (kt + 1 < nk) wait_async<6>(); else wait_async<0>();
#endif
    __syncthreads();
    const int buf = kt & 1;
    v16bf bfrag[4];
#pragma unroll
    for (int b = 0; b < 4; b++) {
      const unsigned short* p = &sB[buf][wn * 64 + b * 16 + l16][16 * half];
      bfrag[b] = ld_frag(p, p + 8);
    }
#pragma unroll
    for (int a = 0; a < 2; a++) {
      const unsigned short* p = &sA[buf][wm * 32 + a * 16 + l16][16 * half];
      v16bf afrag = ld_frag(p, p + 8);
#pragma unroll
      for (int b = 0; b < 4; b++)
        acc[a][b] = wmma_bf16(afrag, bfrag[b], acc[a][b]);
    }
    __syncthreads();
  }

#pragma unroll
  for (int a = 0; a < 2; a++)
#pragma unroll
    for (int b = 0; b < 4; b++)
#pragma unroll
      for (int r = 0; r < 8; r++) {
        int m = bm * 64  + wm * 32 + a * 16 + r + 8 * half;
        int n = bn * 128 + wn * 64 + b * 16 + l16;
        size_t o = (size_t)m * N + n;
        float v = acc[a][b][r];
        if (EPI == 1) {
          Cb[o] = f2bf(v);
        } else if (EPI == 2) {
          Cf[o] = resid[o] + v + bias[n];
        } else if (EPI == 3) {
          Cb[o] = f2bf(gelu_exact(v + bias[n]));
        } else if (EPI == 4) {
          Cf[o] = Cf[o] + scale[(size_t)m * scale_stride] * (v + bias[n]);
        } else {
          Cf[o] = v + bias[n];
        }
      }
}

// ---------------------------------------------------------------------------
// Attention: one workgroup per (b,h); 8 waves, wave w owns query rows
// [16w, 16w+16).  qkv layout: [BT][384] bf16, cols: q = h*32+e, k = 128+...,
// v = 256+...  Scores via WMMA (K-dim = HS = 32 -> single WMMA per tile),
// causal softmax in f32 LDS, P packed bf16 in place, then O = P @ V.
// ---------------------------------------------------------------------------
__global__ __launch_bounds__(256)
void attn_kernel(const unsigned short* __restrict__ qkv,
                 unsigned short* __restrict__ obuf)
{
  __shared__ __align__(16) unsigned short sQ[128][32];
  __shared__ __align__(16) unsigned short sK[128][32];
  __shared__ __align__(16) unsigned short sVt[32][128];
  __shared__ __align__(16) float sS[8][16][128];

  const int bh = blockIdx.x;
  const int b = bh >> 2, h = bh & 3;
  const int tid = threadIdx.x;
  const int w = tid >> 5, lane = tid & 31;
  const int half = lane >> 4, l16 = lane & 15;
  const unsigned short* base = qkv + (size_t)b * TT * 384 + h * 32;

  { // load Q, K rows; V transposed
    int t = tid >> 1;
    int c = (tid & 1) << 4;
    const unsigned short* qp = base + (size_t)t * 384 + c;
    *(uint4*)&sQ[t][c]     = *(const uint4*)qp;
    *(uint4*)&sQ[t][c + 8] = *(const uint4*)(qp + 8);
    const unsigned short* kp = base + 128 + (size_t)t * 384 + c;
    *(uint4*)&sK[t][c]     = *(const uint4*)kp;
    *(uint4*)&sK[t][c + 8] = *(const uint4*)(kp + 8);
    const unsigned short* vp = base + 256 + (size_t)t * 384 + c;
    __align__(16) unsigned short tmp[16];
    *(uint4*)tmp       = *(const uint4*)vp;
    *(uint4*)(tmp + 8) = *(const uint4*)(vp + 8);
#pragma unroll
    for (int j = 0; j < 16; j++) sVt[c + j][t] = tmp[j];
  }
  __syncthreads();

  // --- scores S = QK^T / sqrt(HS), causal mask (wave-uniform loop => EXEC full)
  const unsigned short* qa = &sQ[w * 16 + l16][16 * half];
  v16bf afrag = ld_frag(qa, qa + 8);
  const v8f vz = {0.f,0.f,0.f,0.f,0.f,0.f,0.f,0.f};
  for (int j = 0; j <= w; j++) {
    const unsigned short* kb = &sK[j * 16 + l16][16 * half];
    v16bf bfrag = ld_frag(kb, kb + 8);
    v8f s = wmma_bf16(afrag, bfrag, vz);
#pragma unroll
    for (int r = 0; r < 8; r++) {
      int ml = r + 8 * half;
      int t  = w * 16 + ml;
      int sc = j * 16 + l16;
      float val = s[r] * 0.17677669529663687f;
      sS[w][ml][sc] = (sc <= t) ? val : -1e30f;
    }
  }

  // --- row-wise softmax; write P bf16 in place over the f32 rows (stride 256 u16)
  unsigned short* Pw = (unsigned short*)&sS[w][0][0];
  const int ncols = (w + 1) * 16;
  for (int r = 0; r < 16; r++) {
    float* row = sS[w][r];
    float mx = -1e30f;
    for (int c = lane; c < ncols; c += 32) mx = fmaxf(mx, row[c]);
#pragma unroll
    for (int o = 16; o > 0; o >>= 1) mx = fmaxf(mx, __shfl_xor(mx, o, 32));
    float sum = 0.f;
    for (int c = lane; c < ncols; c += 32) { float e = __expf(row[c] - mx); row[c] = e; sum += e; }
#pragma unroll
    for (int o = 16; o > 0; o >>= 1) sum += __shfl_xor(sum, o, 32);
    float inv = 1.f / sum;
    float tv[4];
#pragma unroll
    for (int i = 0; i < 4; i++) {
      int c = lane + 32 * i;
      tv[i] = (c < ncols) ? row[c] * inv : 0.f;
    }
#pragma unroll
    for (int i = 0; i < 4; i++) Pw[r * 256 + lane + 32 * i] = f2bf(tv[i]);
  }

  // --- O = P @ V  (K = 128 -> 4 WMMA k-steps; 2 N-tiles of 16)
  v8f oacc[2] = {vz, vz};
#pragma unroll
  for (int ks = 0; ks < 4; ks++) {
    const unsigned short* pa = Pw + l16 * 256 + ks * 32 + 16 * half;
    v16bf af = ld_frag(pa, pa + 8);
#pragma unroll
    for (int nb = 0; nb < 2; nb++) {
      const unsigned short* vb = &sVt[nb * 16 + l16][ks * 32 + 16 * half];
      v16bf bf_ = ld_frag(vb, vb + 8);
      oacc[nb] = wmma_bf16(af, bf_, oacc[nb]);
    }
  }
#pragma unroll
  for (int nb = 0; nb < 2; nb++)
#pragma unroll
    for (int r = 0; r < 8; r++) {
      int t = w * 16 + r + 8 * half;
      int e = h * 32 + nb * 16 + l16;
      obuf[((size_t)b * TT + t) * DD + e] = f2bf(oacc[nb][r]);
    }
}

// --------------------------- small kernels ---------------------------------
__global__ __launch_bounds__(256)
void embed_kernel(const int* __restrict__ idx, const float* __restrict__ tok,
                  const float* __restrict__ pos, float* __restrict__ x)
{
  size_t i = (size_t)blockIdx.x * 256 + threadIdx.x;
  int d = (int)(i & (DD - 1));
  size_t row = i >> 7;
  int t = (int)(row & (TT - 1));
  x[i] = tok[(size_t)idx[row] * DD + d] + pos[t * DD + d];
}

__global__ __launch_bounds__(256)
void ln_kernel(const float* __restrict__ xin, const float* __restrict__ g,
               const float* __restrict__ bta, unsigned short* __restrict__ hout)
{
  int lane = threadIdx.x & 31;
  size_t row = (size_t)blockIdx.x * 8 + (threadIdx.x >> 5);
  const float* r = xin + row * DD;
  float v[4]; float s = 0.f;
#pragma unroll
  for (int i = 0; i < 4; i++) { v[i] = r[lane + 32 * i]; s += v[i]; }
#pragma unroll
  for (int o = 16; o > 0; o >>= 1) s += __shfl_xor(s, o, 32);
  float mu = s * (1.f / 128.f);
  float q = 0.f;
#pragma unroll
  for (int i = 0; i < 4; i++) { float d = v[i] - mu; q += d * d; }
#pragma unroll
  for (int o = 16; o > 0; o >>= 1) q += __shfl_xor(q, o, 32);
  float rstd = rsqrtf(q * (1.f / 128.f) + 1e-5f);
#pragma unroll
  for (int i = 0; i < 4; i++) {
    int c = lane + 32 * i;
    hout[row * DD + c] = f2bf((v[i] - mu) * rstd * g[c] + bta[c]);
  }
}

__global__ __launch_bounds__(128)
void gate_kernel(const unsigned short* __restrict__ hbf, const float* __restrict__ gw,
                 const float* __restrict__ gb, float* __restrict__ dw,
                 float* __restrict__ acc)
{
  __shared__ float sacc[16];
  int tid = threadIdx.x;
  if (tid < 16) sacc[tid] = 0.f;
  __syncthreads();
  size_t tok = (size_t)blockIdx.x * 128 + tid;
  const unsigned short* hr = hbf + tok * DD;
  float logit[EE];
#pragma unroll
  for (int e = 0; e < EE; e++) logit[e] = gb[e];
  for (int d = 0; d < DD; d++) {
    float hv = bf2f(hr[d]);
#pragma unroll
    for (int e = 0; e < EE; e++) logit[e] += hv * gw[d * EE + e];
  }
  float mx = logit[0];
#pragma unroll
  for (int e = 1; e < EE; e++) mx = fmaxf(mx, logit[e]);
  float p[EE]; float s = 0.f;
#pragma unroll
  for (int e = 0; e < EE; e++) { p[e] = __expf(logit[e] - mx); s += p[e]; }
  float invs = 1.f / s;
#pragma unroll
  for (int e = 0; e < EE; e++) p[e] *= invs;
  int i0 = 0;
#pragma unroll
  for (int e = 1; e < EE; e++) if (p[e] > p[i0]) i0 = e;
  int i1 = (i0 == 0) ? 1 : 0;
#pragma unroll
  for (int e = 0; e < EE; e++) if (e != i0 && p[e] > p[i1]) i1 = e;
  float wn = 1.f / (p[i0] + p[i1]);
#pragma unroll
  for (int e = 0; e < EE; e++)
    dw[tok * EE + e] = (e == i0) ? p[i0] * wn : ((e == i1) ? p[i1] * wn : 0.f);
#pragma unroll
  for (int e = 0; e < EE; e++) atomicAdd(&sacc[e], p[e]);
  atomicAdd(&sacc[8 + i0], 1.f);
  atomicAdd(&sacc[8 + i1], 1.f);
  __syncthreads();
  if (tid < 16) atomicAdd(&acc[tid], sacc[tid]);
}

__global__ void aux_kernel(const float* __restrict__ acc, float* __restrict__ aux)
{
  if (threadIdx.x == 0) {
    float s = 0.f;
#pragma unroll
    for (int e = 0; e < EE; e++)
      s += (acc[8 + e] * (1.f / (float)(BT * KTOP))) * (acc[e] * (1.f / (float)BT));
    aux[0] += (float)EE * s;
  }
}

__global__ __launch_bounds__(256)
void loss_kernel(const float* __restrict__ logits, const int* __restrict__ tgt,
                 float* __restrict__ loss)
{
  int lane = threadIdx.x & 31;
  size_t row = (size_t)blockIdx.x * 8 + (threadIdx.x >> 5);
  const float* r = logits + row * VV;
  float v[8]; float mx = -3.4e38f;
#pragma unroll
  for (int i = 0; i < 8; i++) { v[i] = r[lane + 32 * i]; mx = fmaxf(mx, v[i]); }
#pragma unroll
  for (int o = 16; o > 0; o >>= 1) mx = fmaxf(mx, __shfl_xor(mx, o, 32));
  float s = 0.f;
#pragma unroll
  for (int i = 0; i < 8; i++) s += __expf(v[i] - mx);
#pragma unroll
  for (int o = 16; o > 0; o >>= 1) s += __shfl_xor(s, o, 32);
  if (lane == 0) {
    float tl = r[tgt[row]];
    float logp = tl - mx - __logf(s);
    atomicAdd(loss, -logp * (1.f / (float)BT));
  }
}

// Transpose-convert: in[b][k][n] (f32, [batch][K][N]) -> out[b][n][k] (bf16)
__global__ void conv_bf16_t(const float* __restrict__ in, unsigned short* __restrict__ o,
                            int Kd, int N, size_t total)
{
  size_t i = (size_t)blockIdx.x * 256 + threadIdx.x;
  if (i >= total) return;
  size_t per = (size_t)Kd * N;
  size_t b = i / per, rem = i % per;
  int n = (int)(rem / Kd), k = (int)(rem % Kd);
  o[i] = f2bf(in[b * per + (size_t)k * N + n]);
}

// Wq/Wk/Wv (L,H,D,HS) -> fused TRANSPOSED [L][384][D] bf16, rows = {q|k|v} n = h*32+e
__global__ void build_wqkv(const float* __restrict__ Wq, const float* __restrict__ Wk,
                           const float* __restrict__ Wv, unsigned short* __restrict__ o, size_t n)
{
  size_t i = (size_t)blockIdx.x * 256 + threadIdx.x;
  if (i >= n) return;
  int d   = (int)(i % DD);
  int col = (int)((i / DD) % (3 * DD));
  int l   = (int)(i / ((size_t)DD * 3 * DD));
  float v;
  if (col < DD) {
    int hh = col >> 5, e = col & 31;
    v = Wq[(((size_t)(l * HH + hh) * DD) + d) * HSZ + e];
  } else if (col < 2 * DD) {
    int c = col - DD; int hh = c >> 5, e = c & 31;
    v = Wk[(((size_t)(l * HH + hh) * DD) + d) * HSZ + e];
  } else {
    int c = col - 2 * DD; int hh = c >> 5, e = c & 31;
    v = Wv[(((size_t)(l * HH + hh) * DD) + d) * HSZ + e];
  }
  o[i] = f2bf(v);
}

// ---------------------------------------------------------------------------
extern "C" void kernel_launch(void* const* d_in, const int* in_sizes, int n_in,
                              void* d_out, int out_size, void* d_ws, size_t ws_size,
                              hipStream_t stream)
{
  (void)in_sizes; (void)n_in; (void)out_size; (void)ws_size;
  const int*   idx     = (const int*)d_in[0];
  const int*   targets = (const int*)d_in[1];
  const float* tok_emb = (const float*)d_in[2];
  const float* pos_emb = (const float*)d_in[3];
  const float* ln1_g   = (const float*)d_in[4];
  const float* ln1_b   = (const float*)d_in[5];
  const float* Wq      = (const float*)d_in[6];
  const float* Wk      = (const float*)d_in[7];
  const float* Wv      = (const float*)d_in[8];
  const float* Wo      = (const float*)d_in[9];
  const float* bo      = (const float*)d_in[10];
  const float* ln2_g   = (const float*)d_in[11];
  const float* ln2_b   = (const float*)d_in[12];
  const float* gate_W  = (const float*)d_in[13];
  const float* gate_b  = (const float*)d_in[14];
  const float* eW1     = (const float*)d_in[15];
  const float* eb1     = (const float*)d_in[16];
  const float* eW2     = (const float*)d_in[17];
  const float* eb2     = (const float*)d_in[18];
  const float* lnf_g   = (const float*)d_in[19];
  const float* lnf_b   = (const float*)d_in[20];
  const float* lm_W    = (const float*)d_in[21];
  const float* lm_b    = (const float*)d_in[22];
  float* out = (float*)d_out;

  char* ws = (char*)d_ws;
  size_t off = 0;
  auto alloc = [&](size_t bytes) -> void* {
    void* p = ws + off; off += (bytes + 255) & ~(size_t)255; return p;
  };
  float*          x      = (float*)alloc((size_t)BT * DD * 4);
  unsigned short* h_bf   = (unsigned short*)alloc((size_t)BT * DD * 2);
  unsigned short* qkv_bf = (unsigned short*)alloc((size_t)BT * 3 * DD * 2);
  unsigned short* o_bf   = (unsigned short*)alloc((size_t)BT * DD * 2);
  unsigned short* he_bf  = (unsigned short*)alloc((size_t)BT * FF * 2);
  float*          dw     = (float*)alloc((size_t)BT * EE * 4);
  unsigned short* wqkv   = (unsigned short*)alloc((size_t)LL * DD * 3 * DD * 2);
  unsigned short* wo_bf  = (unsigned short*)alloc((size_t)LL * DD * DD * 2);
  unsigned short* w1_bf  = (unsigned short*)alloc((size_t)LL * EE * DD * FF * 2);
  unsigned short* w2_bf  = (unsigned short*)alloc((size_t)LL * EE * FF * DD * 2);
  unsigned short* lm_bf  = (unsigned short*)alloc((size_t)DD * VV * 2);
  float*          gacc   = (float*)alloc(64);

  const size_t LOGITS = (size_t)BT * VV;
  (void)hipMemsetAsync(out + LOGITS, 0, 2 * sizeof(float), stream);

  { // weight conversions (bf16, pre-transposed to [N][K] for the GEMM B side)
    size_t n;
    n = (size_t)LL * DD * DD;      // Wo (L,D,D): K=D rows, N=D cols
    conv_bf16_t<<<(unsigned)((n + 255) / 256), 256, 0, stream>>>(Wo, wo_bf, DD, DD, n);
    n = (size_t)LL * EE * DD * FF; // eW1 (L,E,D,F): K=D, N=F
    conv_bf16_t<<<(unsigned)((n + 255) / 256), 256, 0, stream>>>(eW1, w1_bf, DD, FF, n);
    n = (size_t)LL * EE * FF * DD; // eW2 (L,E,F,D): K=F, N=D
    conv_bf16_t<<<(unsigned)((n + 255) / 256), 256, 0, stream>>>(eW2, w2_bf, FF, DD, n);
    n = (size_t)DD * VV;           // lm_W (D,V): K=D, N=V
    conv_bf16_t<<<(unsigned)((n + 255) / 256), 256, 0, stream>>>(lm_W, lm_bf, DD, VV, n);
    n = (size_t)LL * 3 * DD * DD;
    build_wqkv<<<(unsigned)((n + 255) / 256), 256, 0, stream>>>(Wq, Wk, Wv, wqkv, n);
  }

  embed_kernel<<<(BT * DD) / 256, 256, 0, stream>>>(idx, tok_emb, pos_emb, x);

  for (int l = 0; l < LL; l++) {
    ln_kernel<<<BT / 8, 256, 0, stream>>>(x, ln1_g + l * DD, ln1_b + l * DD, h_bf);
    gemm_bf16_kernel<1><<<dim3(BT / 64, (3 * DD) / 128), 128, 0, stream>>>(
        h_bf, wqkv + (size_t)l * DD * 3 * DD, nullptr, nullptr, 0,
        nullptr, qkv_bf, nullptr, BT, 3 * DD, DD);
    attn_kernel<<<BB * HH, 256, 0, stream>>>(qkv_bf, o_bf);
    gemm_bf16_kernel<2><<<dim3(BT / 64, DD / 128), 128, 0, stream>>>(
        o_bf, wo_bf + (size_t)l * DD * DD, bo + l * DD, nullptr, 0,
        x, nullptr, x, BT, DD, DD);
    ln_kernel<<<BT / 8, 256, 0, stream>>>(x, ln2_g + l * DD, ln2_b + l * DD, h_bf);
    (void)hipMemsetAsync(gacc, 0, 64, stream);
    gate_kernel<<<BT / 128, 128, 0, stream>>>(
        h_bf, gate_W + (size_t)l * DD * EE, gate_b + l * EE, dw, gacc);
    aux_kernel<<<1, 32, 0, stream>>>(gacc, out + LOGITS + 1);
    for (int e = 0; e < EE; e++) {
      const size_t we = (size_t)(l * EE + e);
      gemm_bf16_kernel<3><<<dim3(BT / 64, FF / 128), 128, 0, stream>>>(
          h_bf, w1_bf + we * DD * FF, eb1 + we * FF, nullptr, 0,
          nullptr, he_bf, nullptr, BT, FF, DD);
      gemm_bf16_kernel<4><<<dim3(BT / 64, DD / 128), 128, 0, stream>>>(
          he_bf, w2_bf + we * FF * DD, eb2 + we * DD, dw + e, EE,
          x, nullptr, nullptr, BT, DD, FF);
    }
  }

  ln_kernel<<<BT / 8, 256, 0, stream>>>(x, lnf_g, lnf_b, h_bf);
  gemm_bf16_kernel<5><<<dim3(BT / 64, VV / 128), 128, 0, stream>>>(
      h_bf, lm_bf, lm_b, nullptr, 0, out, nullptr, nullptr, BT, VV, DD);
  loss_kernel<<<BT / 8, 256, 0, stream>>>(out, targets, out + LOGITS);
}
